// CustomLM_35759897707218
// MI455X (gfx1250) — matrix operand order repaired and checked by reading, not compiled, not run on hardware
//
#include <hip/hip_runtime.h>
#include <math.h>

// ---- model dims ----
#define V_  50257
#define P_  1024
#define D_  1024
#define H_  16
#define L_  24
#define FF_ 4096
#define HD_ 64
#define B_  2
#define T_  1024
#define BT_ (B_*T_)
#define EPS_ 1e-5f

typedef __attribute__((ext_vector_type(16))) __bf16 v16bf;
typedef __attribute__((ext_vector_type(8)))  float  v8f;

// ---- GEMM tiling ----
#define BM  128
#define BK  32
#define PAD 4   // row stride 36 ushorts = 72B -> 8B aligned for ds_store_b64

__device__ __forceinline__ unsigned short f2bf(float f) {
    unsigned int u = __float_as_uint(f);
    u += 0x7FFFu + ((u >> 16) & 1u);   // round to nearest even
    return (unsigned short)(u >> 16);
}
__device__ __forceinline__ unsigned int pack2(float a, float b) {
    return (unsigned int)f2bf(a) | ((unsigned int)f2bf(b) << 16);
}

union Frag {
    unsigned int d[8];
    v16bf v;
};

// Generic C = act(alpha * A@B + bias + res), bf16 WMMA core, f32 in/out.
// A: [M,K] row-major (f32).  B: [K,N] row-major, or [N,K] row-major if transB.
// Block tile BM x BNv, 8 waves (4 in M x 2 in N); wave tile 32 x (BNv/2).
// Double-buffered LDS staging, one barrier per K-step.
template<int BNv>
__global__ __launch_bounds__(256) void gemm_bf16_wmma(
    const float* __restrict__ A, const float* __restrict__ B, float* __restrict__ C,
    const float* __restrict__ bias, const float* __restrict__ res,
    int M, int N, int K, int lda, int ldb, int ldc,
    int transB, float alpha, int act,
    long long sAo, long long sAi, long long sBo, long long sBi,
    long long sCo, long long sCi, int innerCount)
{
    constexpr int WNT = BNv / 32;                 // 16-wide N tiles per wave (2 or 4)
    __shared__ unsigned short As[2][BM][BK + PAD];
    __shared__ unsigned short Bs[2][BNv][BK + PAD];

    const int z  = blockIdx.z;
    const int zo = z / innerCount;
    const int zi = z % innerCount;
    A += (long long)zo * sAo + (long long)zi * sAi;
    B += (long long)zo * sBo + (long long)zi * sBi;
    const long long cbase = (long long)zo * sCo + (long long)zi * sCi;
    C += cbase;
    if (res) res += cbase;

    const int tid   = threadIdx.x;
    const int lane  = tid & 31;
    const int wave  = tid >> 5;
    const int waveM = wave >> 1;       // 0..3
    const int waveN = wave & 1;        // 0..1
    const int m0 = blockIdx.y * BM;
    const int n0 = blockIdx.x * BNv;

    const bool fullMN = (m0 + BM <= M) && (n0 + BNv <= N) &&
                        ((lda & 3) == 0) && ((ldb & 3) == 0) && ((K & (BK - 1)) == 0);

    // ---- staging helpers ----
    auto stage_fast = [&](int k0, int sbuf) {
        #pragma unroll
        for (int i = 0; i < 4; ++i) {
            int j  = tid + 256 * i;        // 0..1023
            int r  = j >> 3;               // 0..127
            int kq = (j & 7) << 2;         // 0,4,..,28
            const float4 vv = *(const float4*)(A + (long long)(m0 + r) * lda + (k0 + kq));
            uint2 p; p.x = pack2(vv.x, vv.y); p.y = pack2(vv.z, vv.w);
            *(uint2*)&As[sbuf][r][kq] = p;
        }
        if (transB) {
            #pragma unroll
            for (int i = 0; i < BNv / 32; ++i) {
                int j  = tid + 256 * i;    // 0..BNv*8-1
                int n  = j >> 3;           // 0..BNv-1
                int kq = (j & 7) << 2;
                const float4 vv = *(const float4*)(B + (long long)(n0 + n) * ldb + (k0 + kq));
                uint2 p; p.x = pack2(vv.x, vv.y); p.y = pack2(vv.z, vv.w);
                *(uint2*)&Bs[sbuf][n][kq] = p;
            }
        } else {
            constexpr int F4R = BNv / 4;   // float4s per K-row
            #pragma unroll
            for (int i = 0; i < BNv / 32; ++i) {
                int j  = tid + 256 * i;
                int k  = j / F4R;          // 0..31
                int nq = (j % F4R) << 2;
                const float4 vv = *(const float4*)(B + (long long)(k0 + k) * ldb + (n0 + nq));
                Bs[sbuf][nq + 0][k] = f2bf(vv.x);
                Bs[sbuf][nq + 1][k] = f2bf(vv.y);
                Bs[sbuf][nq + 2][k] = f2bf(vv.z);
                Bs[sbuf][nq + 3][k] = f2bf(vv.w);
            }
        }
        // prefetch two slabs ahead (global_prefetch_b8)
        int kp = k0 + BK;
        if (kp < K) {
            int pr = tid >> 1;
            __builtin_prefetch(A + (long long)(m0 + pr) * lda + kp, 0, 3);
            if (tid < BNv) {
                const float* bp = transB
                    ? (B + (long long)(n0 + tid) * ldb + kp)
                    : (B + (long long)kp * ldb + (n0 + tid));
                __builtin_prefetch(bp, 0, 3);
            }
        }
    };

    auto stage_slow = [&](int k0, int sbuf) {
        #pragma unroll
        for (int i = 0; i < (BM * BK) / 256; ++i) {
            int e = tid + 256 * i;
            int r = e >> 5;
            int k = e & 31;
            int gr = m0 + r, gk = k0 + k;
            float v = 0.f;
            if (gr < M && gk < K) v = A[(long long)gr * lda + gk];
            As[sbuf][r][k] = f2bf(v);
        }
        if (transB) {
            #pragma unroll
            for (int i = 0; i < (BNv * BK) / 256; ++i) {
                int e = tid + 256 * i;
                int n = e >> 5;
                int k = e & 31;
                int gn = n0 + n, gk = k0 + k;
                float v = 0.f;
                if (gn < N && gk < K) v = B[(long long)gn * ldb + gk];
                Bs[sbuf][n][k] = f2bf(v);
            }
        } else {
            #pragma unroll
            for (int i = 0; i < (BNv * BK) / 256; ++i) {
                int e = tid + 256 * i;
                int n = e % BNv;
                int k = e / BNv;
                int gn = n0 + n, gk = k0 + k;
                float v = 0.f;
                if (gn < N && gk < K) v = B[(long long)gk * ldb + gn];
                Bs[sbuf][n][k] = f2bf(v);
            }
        }
    };

    v8f acc[2][WNT] = {};

    // fragment lane geometry (ISA VGPR layouts)
    const int rowA   = lane & 15;
    const int khalfA = (lane >> 4) << 3;   // 0 | 8
    const int colB   = lane & 15;
    const int ktopB  = (lane >> 4) << 3;   // 0 | 8

    if (fullMN) stage_fast(0, 0); else stage_slow(0, 0);

    int buf = 0;
    for (int k0 = 0; k0 < K; k0 += BK) {
        __syncthreads();
        int kn = k0 + BK;
        if (kn < K) {
            if (fullMN) stage_fast(kn, buf ^ 1); else stage_slow(kn, buf ^ 1);
        }

        Frag af[2], bfr[WNT];
        #pragma unroll
        for (int wm = 0; wm < 2; ++wm) {
            int rb = waveM * 32 + wm * 16 + rowA;
            #pragma unroll
            for (int v = 0; v < 8; ++v) {
                int kk = ((v & 4) << 2) + khalfA + ((v & 3) << 1);
                af[wm].d[v] = *(const unsigned int*)&As[buf][rb][kk];
            }
        }
        #pragma unroll
        for (int wn = 0; wn < WNT; ++wn) {
            int cb = waveN * (16 * WNT) + wn * 16 + colB;
            #pragma unroll
            for (int v = 0; v < 8; ++v) {
                bfr[wn].d[v] = *(const unsigned int*)&Bs[buf][cb][((v & 4) << 2) + ktopB + ((v & 3) << 1)];
            }
        }
        #pragma unroll
        for (int wm = 0; wm < 2; ++wm)
            #pragma unroll
            for (int wn = 0; wn < WNT; ++wn)
                acc[wm][wn] = __builtin_amdgcn_wmma_f32_16x16x32_bf16(
                    false, af[wm].v, false, bfr[wn].v,
                    (short)0, acc[wm][wn], false, false);
        buf ^= 1;
    }

    // ---- epilogue: C/D layout — VGPR r: M=r (lanes 0-15) / M=8+r (lanes 16-31), N=lane&15 ----
    const int rsel = (lane >> 4) << 3;
    const int cofs = lane & 15;
    if (fullMN) {
        #pragma unroll
        for (int wm = 0; wm < 2; ++wm) {
            #pragma unroll
            for (int wn = 0; wn < WNT; ++wn) {
                int gr0 = m0 + waveM * 32 + wm * 16 + rsel;
                int gc  = n0 + waveN * (16 * WNT) + wn * 16 + cofs;
                float bsv = bias ? bias[gc] : 0.f;
                #pragma unroll
                for (int r = 0; r < 8; ++r) {
                    int gr = gr0 + r;
                    float v = alpha * acc[wm][wn][r] + bsv;
                    if (res)  v += res[(long long)gr * ldc + gc];
                    if (act == 1) v = 0.5f * v * (1.0f + erff(v * 0.7071067811865476f));
                    C[(long long)gr * ldc + gc] = v;
                }
            }
        }
    } else {
        #pragma unroll
        for (int wm = 0; wm < 2; ++wm) {
            #pragma unroll
            for (int wn = 0; wn < WNT; ++wn) {
                int gr0 = m0 + waveM * 32 + wm * 16 + rsel;
                int gc  = n0 + waveN * (16 * WNT) + wn * 16 + cofs;
                if (gc >= N) continue;
                float bsv = bias ? bias[gc] : 0.f;
                #pragma unroll
                for (int r = 0; r < 8; ++r) {
                    int gr = gr0 + r;
                    if (gr >= M) continue;
                    float v = alpha * acc[wm][wn][r] + bsv;
                    if (res)  v += res[(long long)gr * ldc + gc];
                    if (act == 1) v = 0.5f * v * (1.0f + erff(v * 0.7071067811865476f));
                    C[(long long)gr * ldc + gc] = v;
                }
            }
        }
    }
}

// ---- block reductions (blockDim == 256) ----
__device__ __forceinline__ float blk_sum(float v, float* red) {
    int tid = threadIdx.x;
    red[tid] = v; __syncthreads();
    #pragma unroll
    for (int s = 128; s > 0; s >>= 1) {
        if (tid < s) red[tid] += red[tid + s];
        __syncthreads();
    }
    float r = red[0]; __syncthreads();
    return r;
}
__device__ __forceinline__ float blk_max(float v, float* red) {
    int tid = threadIdx.x;
    red[tid] = v; __syncthreads();
    #pragma unroll
    for (int s = 128; s > 0; s >>= 1) {
        if (tid < s) red[tid] = fmaxf(red[tid], red[tid + s]);
        __syncthreads();
    }
    float r = red[0]; __syncthreads();
    return r;
}

// ---- embedding: x[b,t,:] = wte[id] + wpe[t] ----
__global__ __launch_bounds__(256) void embed_kernel(
    const int* __restrict__ ids, const float* __restrict__ wte,
    const float* __restrict__ wpe, float* __restrict__ x)
{
    long long bt = blockIdx.x;
    int t  = (int)(bt % T_);
    int id = ids[bt];
    const float* we = wte + (long long)id * D_;
    const float* wp = wpe + (long long)t * D_;
    float* xr = x + bt * D_;
    for (int i = threadIdx.x; i < D_; i += 256) xr[i] = we[i] + wp[i];
}

// ---- LayerNorm per row of D_ ----
__global__ __launch_bounds__(256) void ln_kernel(
    const float* __restrict__ x, const float* __restrict__ g,
    const float* __restrict__ b, float* __restrict__ o)
{
    __shared__ float red[256];
    long long row = blockIdx.x;
    const float* xr = x + row * D_;
    float s = 0.f, sq = 0.f;
    for (int i = threadIdx.x; i < D_; i += 256) { float v = xr[i]; s += v; sq += v * v; }
    float mean = blk_sum(s, red) * (1.0f / D_);
    float var  = blk_sum(sq, red) * (1.0f / D_) - mean * mean;
    float rstd = rsqrtf(var + EPS_);
    float* orow = o + row * D_;
    for (int i = threadIdx.x; i < D_; i += 256)
        orow[i] = (xr[i] - mean) * rstd * g[i] + b[i];
}

// ---- causal softmax: one block per (b,h,q) row of length T_ ----
__global__ __launch_bounds__(256) void softmax_kernel(float* __restrict__ att)
{
    __shared__ float red[256];
    long long rid = blockIdx.x;          // z*T + q
    int q = (int)(rid % T_);
    float* row = att + rid * T_;
    int n = q + 1;
    float mx = -3.0e38f;
    for (int i = threadIdx.x; i < n; i += 256) mx = fmaxf(mx, row[i]);
    float m = blk_max(mx, red);
    float s = 0.f;
    for (int i = threadIdx.x; i < n; i += 256) { float e = expf(row[i] - m); row[i] = e; s += e; }
    float S = blk_sum(s, red);
    float inv = 1.0f / S;
    for (int i = threadIdx.x; i < T_; i += 256) row[i] = (i < n) ? row[i] * inv : 0.f;
}

// ---- per-row NLL over V_ logits (shifted) ----
__global__ __launch_bounds__(256) void nll_kernel(
    const float* __restrict__ logits, const int* __restrict__ labels,
    float* __restrict__ nll, float* __restrict__ vld)
{
    __shared__ float red[256];
    int r = blockIdx.x;                  // 0 .. B*(T-1)-1
    int b = r / (T_ - 1);
    int t = r % (T_ - 1);
    const float* lr = logits + ((long long)b * T_ + t) * (long long)V_;
    int lab = labels[b * T_ + t + 1];
    if (lab == -100) {
        if (threadIdx.x == 0) { nll[r] = 0.f; vld[r] = 0.f; }
        return;
    }
    float mx = -3.0e38f;
    for (int i = threadIdx.x; i < V_; i += 256) mx = fmaxf(mx, lr[i]);
    float m = blk_max(mx, red);
    float s = 0.f;
    for (int i = threadIdx.x; i < V_; i += 256) s += expf(lr[i] - m);
    float S = blk_sum(s, red);
    if (threadIdx.x == 0) {
        nll[r] = logf(S) + m - lr[lab];
        vld[r] = 1.f;
    }
}

__global__ __launch_bounds__(256) void finalize_loss_kernel(
    const float* __restrict__ nll, const float* __restrict__ vld,
    int n, float* __restrict__ loss)
{
    __shared__ float r1[256];
    __shared__ float r2[256];
    float s = 0.f, c = 0.f;
    for (int i = threadIdx.x; i < n; i += 256) { s += nll[i]; c += vld[i]; }
    float S = blk_sum(s, r1);
    float C = blk_sum(c, r2);
    if (threadIdx.x == 0) loss[0] = S / fmaxf(C, 1.0f);
}

extern "C" void kernel_launch(void* const* d_in, const int* in_sizes, int n_in,
                              void* d_out, int out_size, void* d_ws, size_t ws_size,
                              hipStream_t stream)
{
    const int*   ids         = (const int*)  d_in[0];
    const int*   labels      = (const int*)  d_in[1];
    const float* wte         = (const float*)d_in[2];
    const float* wpe         = (const float*)d_in[3];
    const float* attn_w      = (const float*)d_in[4];
    const float* attn_b      = (const float*)d_in[5];
    const float* attn_proj_w = (const float*)d_in[6];
    const float* attn_proj_b = (const float*)d_in[7];
    const float* fc_w        = (const float*)d_in[8];
    const float* fc_b        = (const float*)d_in[9];
    const float* fc_proj_w   = (const float*)d_in[10];
    const float* fc_proj_b   = (const float*)d_in[11];
    const float* ln1_g       = (const float*)d_in[12];
    const float* ln1_b       = (const float*)d_in[13];
    const float* ln2_g       = (const float*)d_in[14];
    const float* ln2_b       = (const float*)d_in[15];
    const float* lnf_g       = (const float*)d_in[16];
    const float* lnf_b       = (const float*)d_in[17];

    float* out    = (float*)d_out;
    float* logits = out;                               // B*T*V
    float* loss   = out + (long long)BT_ * V_;         // 1

    char* ws = (char*)d_ws;
    float* x   = (float*)ws; ws += (long long)BT_ * D_      * 4;
    float* h   = (float*)ws; ws += (long long)BT_ * D_      * 4;
    float* qkv = (float*)ws; ws += (long long)BT_ * 3 * D_  * 4;
    float* y   = (float*)ws; ws += (long long)BT_ * D_      * 4;
    float* mb  = (float*)ws; ws += (long long)BT_ * FF_     * 4;
    float* nll = (float*)ws; ws += (long long)BT_           * 4;
    float* vld = (float*)ws; ws += (long long)BT_           * 4;
    float* att = (float*)ws;                           // B*H*T*T

    auto gemm = [&](const float* A, const float* Bm, float* C,
                    const float* bias, const float* res,
                    int M, int N, int K, int lda, int ldb, int ldc,
                    int transB, float alpha, int act,
                    int nz, int inner,
                    long long sAo, long long sAi, long long sBo, long long sBi,
                    long long sCo, long long sCi) {
        if (N >= 128) {
            dim3 g((unsigned)((N + 127) / 128), (unsigned)((M + BM - 1) / BM), (unsigned)nz);
            gemm_bf16_wmma<128><<<g, 256, 0, stream>>>(A, Bm, C, bias, res,
                                                       M, N, K, lda, ldb, ldc,
                                                       transB, alpha, act,
                                                       sAo, sAi, sBo, sBi, sCo, sCi, inner);
        } else {
            dim3 g((unsigned)((N + 63) / 64), (unsigned)((M + BM - 1) / BM), (unsigned)nz);
            gemm_bf16_wmma<64><<<g, 256, 0, stream>>>(A, Bm, C, bias, res,
                                                      M, N, K, lda, ldb, ldc,
                                                      transB, alpha, act,
                                                      sAo, sAi, sBo, sBi, sCo, sCi, inner);
        }
    };

    const float scale = 0.125f;  // 1/sqrt(HD)

    // embeddings
    embed_kernel<<<BT_, 256, 0, stream>>>(ids, wte, wpe, x);

    for (int l = 0; l < L_; ++l) {
        // ln1
        ln_kernel<<<BT_, 256, 0, stream>>>(x, ln1_g + (long long)l * D_, ln1_b + (long long)l * D_, h);
        // qkv = h @ attn_w + attn_b    [BT, 3D]
        gemm(h, attn_w + (long long)l * D_ * 3 * D_, qkv,
             attn_b + (long long)l * 3 * D_, nullptr,
             BT_, 3 * D_, D_, D_, 3 * D_, 3 * D_, 0, 1.f, 0,
             1, 1, 0, 0, 0, 0, 0, 0);
        // att[b,h] = scale * Q @ K^T   (batched over z = b*H + h)
        gemm(qkv, qkv + D_, att, nullptr, nullptr,
             T_, T_, HD_, 3 * D_, 3 * D_, T_, 1, scale, 0,
             B_ * H_, H_,
             (long long)T_ * 3 * D_, HD_,
             (long long)T_ * 3 * D_, HD_,
             (long long)H_ * T_ * T_, (long long)T_ * T_);
        // causal softmax
        softmax_kernel<<<B_ * H_ * T_, 256, 0, stream>>>(att);
        // y[b,:,h*HD:] = att @ V
        gemm(att, qkv + 2 * D_, y, nullptr, nullptr,
             T_, HD_, T_, T_, 3 * D_, D_, 0, 1.f, 0,
             B_ * H_, H_,
             (long long)H_ * T_ * T_, (long long)T_ * T_,
             (long long)T_ * 3 * D_, HD_,
             (long long)T_ * D_, HD_);
        // x = x + y @ attn_proj_w + attn_proj_b
        gemm(y, attn_proj_w + (long long)l * D_ * D_, x,
             attn_proj_b + (long long)l * D_, x,
             BT_, D_, D_, D_, D_, D_, 0, 1.f, 0,
             1, 1, 0, 0, 0, 0, 0, 0);
        // ln2
        ln_kernel<<<BT_, 256, 0, stream>>>(x, ln2_g + (long long)l * D_, ln2_b + (long long)l * D_, h);
        // m = gelu(h @ fc_w + fc_b)
        gemm(h, fc_w + (long long)l * D_ * FF_, mb,
             fc_b + (long long)l * FF_, nullptr,
             BT_, FF_, D_, D_, FF_, FF_, 0, 1.f, 1,
             1, 1, 0, 0, 0, 0, 0, 0);
        // x = x + m @ fc_proj_w + fc_proj_b
        gemm(mb, fc_proj_w + (long long)l * FF_ * D_, x,
             fc_proj_b + (long long)l * D_, x,
             BT_, D_, FF_, FF_, D_, D_, 0, 1.f, 0,
             1, 1, 0, 0, 0, 0, 0, 0);
    }

    // final LN
    ln_kernel<<<BT_, 256, 0, stream>>>(x, lnf_g, lnf_b, h);
    // logits = h @ wte^T   [BT, V]
    gemm(h, wte, logits, nullptr, nullptr,
         BT_, V_, D_, D_, D_, V_, 1, 1.f, 0,
         1, 1, 0, 0, 0, 0, 0, 0);

    // loss
    nll_kernel<<<B_ * (T_ - 1), 256, 0, stream>>>(logits, labels, nll, vld);
    finalize_loss_kernel<<<1, 256, 0, stream>>>(nll, vld, B_ * (T_ - 1), loss);
}
